// CCM_Model_79396765434379
// MI455X (gfx1250) — matrix operand order repaired and checked by reading, not compiled, not run on hardware
//
#include <hip/hip_runtime.h>
#include <hip/hip_fp16.h>

// ---------------- model dims ----------------
#define BB    32
#define SS    50
#define KTR   32
#define ENT   100
#define REL   200
#define TRIPD 400
#define WEMB  300
#define HENC  256
#define HDEC  256
#define VSZ   30000
#define GINT  100
#define VG    64
#define VE    256
#define DECIN 1556   // 200 + 400 + 400 + 256 + 300

typedef __attribute__((ext_vector_type(16))) _Float16 v16h;
typedef __attribute__((ext_vector_type(8)))  float    v8f;

// ---------------- small utility kernels ----------------
__global__ void k_f32_to_f16(const float* __restrict__ src, _Float16* __restrict__ dst, int n) {
  for (int i = blockIdx.x * blockDim.x + threadIdx.x; i < n; i += gridDim.x * blockDim.x)
    dst[i] = (_Float16)src[i];
}

__global__ void k_zero(float* __restrict__ p, int n) {
  for (int i = blockIdx.x * blockDim.x + threadIdx.x; i < n; i += gridDim.x * blockDim.x)
    p[i] = 0.0f;
}

__device__ __forceinline__ float sigf(float x) { return 1.0f / (1.0f + __expf(-x)); }

// ---------------- phase 1: graph encoder attention -> g1 [B,S,200] ----------------
__global__ __launch_bounds__(128)
void k_graph_att(const float* __restrict__ Er, const float* __restrict__ Ei,
                 const float* __restrict__ gWh, const float* __restrict__ gbh,
                 const float* __restrict__ gWt, const float* __restrict__ gbt,
                 const float* __restrict__ gWr, const float* __restrict__ gbr,
                 float* __restrict__ g1) {
  __shared__ float sEr[4][TRIPD];
  __shared__ float sEi[4][TRIPD];
  __shared__ float beta[KTR];
  __shared__ float alpha[KTR];
  const int bs   = blockIdx.x;              // b*S + s
  const int tid  = threadIdx.x;
  const int wave = tid >> 5, lane = tid & 31;
  const float* er = Er + (size_t)bs * KTR * TRIPD;
  const float* ei = Ei + (size_t)bs * KTR * TRIPD;

  for (int kg = 0; kg < KTR; kg += 4) {
    for (int i = tid; i < 4 * TRIPD; i += 128) {
      int kk = i / TRIPD, d = i % TRIPD;
      sEr[kk][d] = er[(size_t)(kg + kk) * TRIPD + d];
      sEi[kk][d] = ei[(size_t)(kg + kk) * TRIPD + d];
    }
    __syncthreads();
    float accR = 0.f, accI = 0.f;
    for (int g = lane; g < GINT; g += 32) {
      float hr = gbh[g], hi = gbh[GINT + g];
      for (int e = 0; e < ENT; ++e) {
        float wr = gWh[g * ENT + e], wi = gWh[GINT * ENT + g * ENT + e];
        float xr = sEr[wave][e],     xi = sEi[wave][e];
        hr += xr * wr - xi * wi;  hi += xr * wi + xi * wr;
      }
      float tr = gbt[g], ti = gbt[GINT + g];
      for (int e = 0; e < ENT; ++e) {
        float wr = gWt[g * ENT + e], wi = gWt[GINT * ENT + g * ENT + e];
        float xr = sEr[wave][ENT + e], xi = sEi[wave][ENT + e];
        tr += xr * wr - xi * wi;  ti += xr * wi + xi * wr;
      }
      float rr = gbr[g], ri = gbr[GINT + g];
      for (int e = 0; e < REL; ++e) {
        float wr = gWr[g * REL + e], wi = gWr[GINT * REL + g * REL + e];
        float xr = sEr[wave][2 * ENT + e], xi = sEi[wave][2 * ENT + e];
        rr += xr * wr - xi * wi;  ri += xr * wi + xi * wr;
      }
      float qr = fmaxf(hr + tr, 0.f), qi = fmaxf(hi + ti, 0.f);   // crelu
      accR += rr * qr + ri * qi;                                   // rh * conj(q)
      accI += ri * qr - rr * qi;
    }
    for (int off = 16; off; off >>= 1) {
      accR += __shfl_xor(accR, off, 32);
      accI += __shfl_xor(accI, off, 32);
    }
    if (lane == 0) beta[kg + wave] = sqrtf(accR * accR + accI * accI);
    __syncthreads();
  }
  if (wave == 0) {
    float v = beta[lane], m = v;
    for (int off = 16; off; off >>= 1) m = fmaxf(m, __shfl_xor(m, off, 32));
    float e = __expf(v - m), s = e;
    for (int off = 16; off; off >>= 1) s += __shfl_xor(s, off, 32);
    alpha[lane] = e / s;
  }
  __syncthreads();
  for (int d = tid; d < 2 * ENT; d += 128) {
    float sr = 0.f, si = 0.f;
    for (int k = 0; k < KTR; ++k) {
      float a = alpha[k];
      sr += a * er[(size_t)k * TRIPD + d];
      si += a * ei[(size_t)k * TRIPD + d];
    }
    g1[(size_t)bs * 2 * ENT + d] = sqrtf(sr * sr + si * si);
  }
}

// ---------------- phase 2: encoder GRU step (both layers fused) ----------------
__global__ __launch_bounds__(256)
void k_enc_step(const float* __restrict__ wemb, const float* __restrict__ g1,
                const float* __restrict__ h0_in, const float* __restrict__ h1_in,
                float* __restrict__ h0_out, float* __restrict__ h1_out,
                const float* __restrict__ Wih0, const float* __restrict__ Whh0,
                const float* __restrict__ bih0, const float* __restrict__ bhh0,
                const float* __restrict__ Wih1, const float* __restrict__ Whh1,
                const float* __restrict__ bih1, const float* __restrict__ bhh1,
                float* __restrict__ enc_all, int t) {
  __shared__ float xs[WEMB + 2 * ENT];   // 500
  __shared__ float h0s[HENC], nh0s[HENC], h1s[HENC];
  const int b = blockIdx.x, j = threadIdx.x;
  for (int i = j; i < WEMB; i += 256) xs[i] = wemb[((size_t)b * SS + t) * WEMB + i];
  for (int i = j; i < 2 * ENT; i += 256) xs[WEMB + i] = g1[((size_t)b * SS + t) * 2 * ENT + i];
  h0s[j] = h0_in[b * HENC + j];
  h1s[j] = h1_in[b * HENC + j];
  __syncthreads();
  { // layer 0
    float gr = bih0[j], gz = bih0[256 + j], gn = bih0[512 + j];
    for (int i = 0; i < 500; ++i) {
      float x = xs[i];
      gr += x * Wih0[(size_t)j * 500 + i];
      gz += x * Wih0[(size_t)(256 + j) * 500 + i];
      gn += x * Wih0[(size_t)(512 + j) * 500 + i];
    }
    float hr = bhh0[j], hz = bhh0[256 + j], hn = bhh0[512 + j];
    for (int i = 0; i < 256; ++i) {
      float h = h0s[i];
      hr += h * Whh0[j * 256 + i];
      hz += h * Whh0[(256 + j) * 256 + i];
      hn += h * Whh0[(512 + j) * 256 + i];
    }
    float r = sigf(gr + hr), z = sigf(gz + hz);
    float n = tanhf(gn + r * hn);
    float nh = (1.f - z) * n + z * h0s[j];
    nh0s[j] = nh;
    h0_out[b * HENC + j] = nh;
  }
  __syncthreads();
  { // layer 1 (input = new h0)
    float gr = bih1[j], gz = bih1[256 + j], gn = bih1[512 + j];
    for (int i = 0; i < 256; ++i) {
      float x = nh0s[i];
      gr += x * Wih1[j * 256 + i];
      gz += x * Wih1[(256 + j) * 256 + i];
      gn += x * Wih1[(512 + j) * 256 + i];
    }
    float hr = bhh1[j], hz = bhh1[256 + j], hn = bhh1[512 + j];
    for (int i = 0; i < 256; ++i) {
      float h = h1s[i];
      hr += h * Whh1[j * 256 + i];
      hz += h * Whh1[(256 + j) * 256 + i];
      hn += h * Whh1[(512 + j) * 256 + i];
    }
    float r = sigf(gr + hr), z = sigf(gz + hz);
    float n = tanhf(gn + r * hn);
    float nh = (1.f - z) * n + z * h1s[j];
    h1_out[b * HENC + j] = nh;
    enc_all[((size_t)b * SS + t) * HENC + j] = nh;
  }
}

// ---------------- g_top / e_top projections ----------------
__global__ __launch_bounds__(64)
void k_gtop(const float* __restrict__ g1, const float* __restrict__ W, float* __restrict__ out) {
  int bs = blockIdx.x, v = threadIdx.x;
  const float* g = g1 + (size_t)bs * 2 * ENT;
  float acc = 0.f;
  for (int d = 0; d < 2 * ENT; ++d) acc += g[d] * W[v * 2 * ENT + d];
  out[(size_t)bs * VG + v] = acc;
}

__global__ __launch_bounds__(256)
void k_etop(const float* __restrict__ enc, const float* __restrict__ W, float* __restrict__ out) {
  int bs = blockIdx.x, v = threadIdx.x;
  const float* e = enc + (size_t)bs * HENC;
  float acc = 0.f;
  for (int d = 0; d < HENC; ++d) acc += e[d] * W[v * HENC + d];
  out[(size_t)bs * VE + v] = acc;
}

// ---------------- WMMA vocab projection: logits = h[32,256] @ W_word[30000,256]^T ----------------
__global__ __launch_bounds__(256)
void k_logits_wmma(const _Float16* __restrict__ hf16,   // [32,256] row-major
                   const _Float16* __restrict__ Wf16,   // [30000,256] row-major
                   float* __restrict__ logits) {        // [32,30000]
  __shared__ _Float16 hs[32 * 256];                     // 16 KB, shared by 8 waves
  const int tid = threadIdx.x;
  { // cooperative 16B-wide stage of h into LDS
    const float4* src = (const float4*)hf16;
    float4* dst = (float4*)hs;
    for (int i = tid; i < (32 * 256 * 2) / 16; i += 256) dst[i] = src[i];
  }
  __syncthreads();
  const int wave  = tid >> 5, lane = tid & 31;
  const int ntile = blockIdx.x * 8 + wave;              // 16-wide N tile
  if (ntile >= VSZ / 16) return;                        // wave-uniform, EXEC stays full
  const int hi = lane >> 4;                             // half-wave selector
  const int m  = lane & 15;
  const int n  = ntile * 16 + (lane & 15);
  v8f acc0 = {}; v8f acc1 = {};
#pragma unroll
  for (int kk = 0; kk < 8; ++kk) {                      // K = 256 in chunks of 32
    v16h a0, a1, bf;
    const _Float16* hrow0 = hs + m * 256 + kk * 32;
    const _Float16* hrow1 = hs + (m + 16) * 256 + kk * 32;
#pragma unroll
    for (int v = 0; v < 8; ++v) {                       // A: contiguous k = 16*hi + s (matches B)
      int base = 2 * v + hi * 16;
      a0[2 * v]     = hrow0[base];
      a0[2 * v + 1] = hrow0[base + 1];
      a1[2 * v]     = hrow1[base];
      a1[2 * v + 1] = hrow1[base + 1];
    }
    const _Float16* wrow = Wf16 + (size_t)n * 256 + kk * 32 + hi * 16;
    __builtin_prefetch(wrow + 32, 0, 0);                // global_prefetch next K chunk
#pragma unroll
    for (int v = 0; v < 16; ++v) bf[v] = wrow[v];       // 16 contiguous halves -> b128 loads
    acc0 = __builtin_amdgcn_wmma_f32_16x16x32_f16(false, a0, false, bf, (short)0, acc0, false, false);
    acc1 = __builtin_amdgcn_wmma_f32_16x16x32_f16(false, a1, false, bf, (short)0, acc1, false, false);
  }
#pragma unroll
  for (int r = 0; r < 8; ++r) {                         // ISA 32-bit C/D layout
    int row = r + hi * 8;
    logits[(size_t)row * VSZ + n]        = acc0[r];
    logits[(size_t)(row + 16) * VSZ + n] = acc1[r];
  }
}

// ---------------- log_softmax + argmax over V ----------------
__global__ __launch_bounds__(256)
void k_logsoftmax_argmax(const float* __restrict__ logits, float* __restrict__ out,
                         int* __restrict__ idx, int t) {
  __shared__ float smax[8], ssum[8];
  __shared__ int   sarg[8];
  const int b = blockIdx.x, tid = threadIdx.x;
  const int wave = tid >> 5, lane = tid & 31;
  const float* row = logits + (size_t)b * VSZ;
  float m = -1e30f; int arg = 0;
  for (int i = tid; i < VSZ; i += 256) {
    float v = row[i];
    if (v > m) { m = v; arg = i; }
  }
  for (int off = 16; off; off >>= 1) {
    float om = __shfl_xor(m, off, 32);
    int   oa = __shfl_xor(arg, off, 32);
    if (om > m || (om == m && oa < arg)) { m = om; arg = oa; }
  }
  if (lane == 0) { smax[wave] = m; sarg[wave] = arg; }
  __syncthreads();
  if (tid == 0) {
    float mm = smax[0]; int aa = sarg[0];
    for (int w = 1; w < 8; ++w)
      if (smax[w] > mm || (smax[w] == mm && sarg[w] < aa)) { mm = smax[w]; aa = sarg[w]; }
    smax[0] = mm;
    if (idx) idx[b] = aa;
  }
  __syncthreads();
  m = smax[0];
  float s = 0.f;
  for (int i = tid; i < VSZ; i += 256) s += __expf(row[i] - m);
  for (int off = 16; off; off >>= 1) s += __shfl_xor(s, off, 32);
  if (lane == 0) ssum[wave] = s;
  __syncthreads();
  if (tid == 0) { float a = 0.f; for (int w = 0; w < 8; ++w) a += ssum[w]; ssum[0] = a; }
  __syncthreads();
  float lse = m + __logf(ssum[0]);
  if (out) {
    float* o = out + ((size_t)b * SS + t) * VSZ;
    for (int i = tid; i < VSZ; i += 256) o[i] = row[i] - lse;
  }
}

__global__ void k_gather_pw(const float* __restrict__ lookup, const int* __restrict__ idx,
                            float* __restrict__ pw) {
  int b = blockIdx.x;
  int id = idx[b];
  for (int i = threadIdx.x; i < WEMB; i += blockDim.x)
    pw[b * WEMB + i] = lookup[(size_t)id * WEMB + i];
}

// ---------------- decoder attention (graph, VG=64) ----------------
__global__ __launch_bounds__(256)
void k_dec_att_g(const float* __restrict__ h, const float* __restrict__ g_top,
                 const float* __restrict__ g1, const float* __restrict__ Wh,
                 const float* __restrict__ Wm, float* __restrict__ c_g,
                 float* __restrict__ alpha_top) {
  __shared__ float hs[HDEC], hW[VG], alpha[SS];
  const int b = blockIdx.x, tid = threadIdx.x;
  hs[tid] = h[b * HDEC + tid];
  __syncthreads();
  if (tid < VG) {
    float acc = 0.f;
    for (int i = 0; i < HDEC; ++i) acc += hs[i] * Wh[tid * HDEC + i];
    hW[tid] = acc;
  }
  __syncthreads();
  if (tid < SS) {
    float acc = 0.f;
    const float* gt = g_top + ((size_t)b * SS + tid) * VG;
    for (int v = 0; v < VG; ++v) acc += (hW[v] + gt[v]) * Wm[v];
    alpha[tid] = acc;
  }
  __syncthreads();
  if (tid == 0) {
    float m = -1e30f;
    for (int t2 = 0; t2 < SS; ++t2) m = fmaxf(m, alpha[t2]);
    float s = 0.f;
    for (int t2 = 0; t2 < SS; ++t2) { alpha[t2] = __expf(alpha[t2] - m); s += alpha[t2]; }
    float inv = 1.f / s;
    for (int t2 = 0; t2 < SS; ++t2) alpha[t2] *= inv;
  }
  __syncthreads();
  for (int t2 = tid; t2 < SS; t2 += 256) alpha_top[b * SS + t2] = alpha[t2];
  for (int d = tid; d < 2 * ENT; d += 256) {
    float acc = 0.f;
    for (int t2 = 0; t2 < SS; ++t2) acc += alpha[t2] * g1[((size_t)b * SS + t2) * 2 * ENT + d];
    c_g[b * 2 * ENT + d] = acc;
  }
}

// ---------------- decoder attention (encoder, VE=256) ----------------
__global__ __launch_bounds__(256)
void k_dec_att_e(const float* __restrict__ h, const float* __restrict__ e_top,
                 const float* __restrict__ enc_all, const float* __restrict__ Wh,
                 const float* __restrict__ Wm, float* __restrict__ c_e) {
  __shared__ float hs[HDEC], hW[VE], alpha[SS];
  const int b = blockIdx.x, tid = threadIdx.x;
  hs[tid] = h[b * HDEC + tid];
  __syncthreads();
  {
    float acc = 0.f;
    for (int i = 0; i < HDEC; ++i) acc += hs[i] * Wh[tid * HDEC + i];
    hW[tid] = acc;
  }
  __syncthreads();
  if (tid < SS) {
    float acc = 0.f;
    const float* et = e_top + ((size_t)b * SS + tid) * VE;
    for (int v = 0; v < VE; ++v) acc += (hW[v] + et[v]) * Wm[v];
    alpha[tid] = acc;
  }
  __syncthreads();
  if (tid == 0) {
    float m = -1e30f;
    for (int t2 = 0; t2 < SS; ++t2) m = fmaxf(m, alpha[t2]);
    float s = 0.f;
    for (int t2 = 0; t2 < SS; ++t2) { alpha[t2] = __expf(alpha[t2] - m); s += alpha[t2]; }
    float inv = 1.f / s;
    for (int t2 = 0; t2 < SS; ++t2) alpha[t2] *= inv;
  }
  __syncthreads();
  {
    float acc = 0.f;
    for (int t2 = 0; t2 < SS; ++t2) acc += alpha[t2] * enc_all[((size_t)b * SS + t2) * HENC + tid];
    c_e[b * HENC + tid] = acc;
  }
}

// ---------------- inter = conj(clin(h, tW_map, tb_map)) ----------------
__global__ __launch_bounds__(256)
void k_inter(const float* __restrict__ h, const float* __restrict__ tW,
             const float* __restrict__ tb, float* __restrict__ ire, float* __restrict__ iim) {
  __shared__ float hs[HDEC];
  const int b = blockIdx.x, tid = threadIdx.x;
  hs[tid] = h[b * HDEC + tid];
  __syncthreads();
  for (int d = tid; d < TRIPD; d += 256) {
    float ar = tb[d], ai = tb[TRIPD + d];
    for (int i = 0; i < HDEC; ++i) {
      float x = hs[i];
      ar += x * tW[(size_t)d * HDEC + i];
      ai += x * tW[(size_t)TRIPD * HDEC + (size_t)d * HDEC + i];
    }
    ire[b * TRIPD + d] = ar;
    iim[b * TRIPD + d] = -ai;    // conj
  }
}

// ---------------- beta_t + softmax_k, scaled by alpha_top -> a[B,S,K] ----------------
__global__ __launch_bounds__(128)
void k_beta_t(const float* __restrict__ Er, const float* __restrict__ Ei,
              const float* __restrict__ ire, const float* __restrict__ iim,
              const float* __restrict__ alpha_top, float* __restrict__ a_buf) {
  __shared__ float sir[TRIPD], sii[TRIPD], beta[KTR];
  const int bs = blockIdx.x, b = bs / SS;
  const int tid = threadIdx.x, wave = tid >> 5, lane = tid & 31;
  const float* er = Er + (size_t)bs * KTR * TRIPD;
  const float* ei = Ei + (size_t)bs * KTR * TRIPD;
  for (int i = tid; i < TRIPD; i += 128) { sir[i] = ire[b * TRIPD + i]; sii[i] = iim[b * TRIPD + i]; }
  __syncthreads();
  for (int k = wave; k < KTR; k += 4) {
    float ar = 0.f, ai = 0.f;
    for (int d = lane; d < TRIPD; d += 32) {
      float xr = er[(size_t)k * TRIPD + d], xi = ei[(size_t)k * TRIPD + d];
      float yr = sir[d], yi = sii[d];
      ar += xr * yr - xi * yi;
      ai += xr * yi + xi * yr;
    }
    for (int off = 16; off; off >>= 1) {
      ar += __shfl_xor(ar, off, 32);
      ai += __shfl_xor(ai, off, 32);
    }
    if (lane == 0) beta[k] = sqrtf(ar * ar + ai * ai);
  }
  __syncthreads();
  if (wave == 0) {
    float v = beta[lane], m = v;
    for (int off = 16; off; off >>= 1) m = fmaxf(m, __shfl_xor(m, off, 32));
    float e = __expf(v - m), s = e;
    for (int off = 16; off; off >>= 1) s += __shfl_xor(s, off, 32);
    a_buf[(size_t)bs * KTR + lane] = alpha_top[bs] * e / s;
  }
}

// ---------------- c_hier + sub/trip argmax + pk_new ----------------
__global__ __launch_bounds__(256)
void k_chier(const float* __restrict__ Er, const float* __restrict__ Ei,
             const float* __restrict__ a_buf, float* __restrict__ c_hier,
             float* __restrict__ pk_out) {
  __shared__ float sa[SS * KTR];     // 1600
  __shared__ float tmax[SS];
  __shared__ int   targ[SS];
  __shared__ int   ssub, strip;
  const int b = blockIdx.x, tid = threadIdx.x;
  for (int i = tid; i < SS * KTR; i += 256) sa[i] = a_buf[(size_t)b * SS * KTR + i];
  __syncthreads();
  if (tid < SS) {
    float m = -1e30f; int arg = 0;
    for (int k = 0; k < KTR; ++k) {
      float v = sa[tid * KTR + k];
      if (v > m) { m = v; arg = k; }
    }
    tmax[tid] = m; targ[tid] = arg;
  }
  __syncthreads();
  if (tid == 0) {
    float m = -1e30f; int sub = 0;
    for (int t2 = 0; t2 < SS; ++t2) if (tmax[t2] > m) { m = tmax[t2]; sub = t2; }
    ssub = sub; strip = targ[sub];
  }
  __syncthreads();
  const int sub = ssub, trip = strip;
  const float* er = Er + (size_t)b * SS * KTR * TRIPD;
  const float* ei = Ei + (size_t)b * SS * KTR * TRIPD;
  for (int d = tid; d < TRIPD; d += 256) {
    float ar = 0.f, ai = 0.f;
    for (int t2 = 0; t2 < SS; ++t2)
      for (int k = 0; k < KTR; ++k) {
        float a = sa[t2 * KTR + k];
        ar += a * er[((size_t)t2 * KTR + k) * TRIPD + d];
        ai += a * ei[((size_t)t2 * KTR + k) * TRIPD + d];
      }
    c_hier[b * TRIPD + d] = sqrtf(ar * ar + ai * ai);
    float xr = er[((size_t)sub * KTR + trip) * TRIPD + d];
    float xi = ei[((size_t)sub * KTR + trip) * TRIPD + d];
    pk_out[b * TRIPD + d] = sqrtf(xr * xr + xi * xi);
  }
}

// ---------------- decoder input concat ----------------
__global__ __launch_bounds__(256)
void k_build_x(const float* __restrict__ c_g, const float* __restrict__ c_hier,
               const float* __restrict__ pk, const float* __restrict__ c_e,
               const float* __restrict__ pw, float* __restrict__ x) {
  const int b = blockIdx.x, tid = threadIdx.x;
  float* xb = x + (size_t)b * DECIN;
  for (int i = tid; i < 200; i += 256) xb[i]        = c_g[b * 200 + i];
  for (int i = tid; i < 400; i += 256) xb[200 + i]  = c_hier[b * 400 + i];
  for (int i = tid; i < 400; i += 256) xb[600 + i]  = pk[b * 400 + i];
  for (int i = tid; i < 256; i += 256) xb[1000 + i] = c_e[b * 256 + i];
  for (int i = tid; i < 300; i += 256) xb[1256 + i] = pw[b * 300 + i];
}

// ---------------- decoder GRU (layer1 reads OLD h0 -> both layers independent) ----------------
__global__ __launch_bounds__(256)
void k_dec_gru(const float* __restrict__ x,
               const float* __restrict__ h0_in, const float* __restrict__ h1_in,
               float* __restrict__ h0_out, float* __restrict__ h1_out,
               const float* __restrict__ Wih0, const float* __restrict__ Whh0,
               const float* __restrict__ bih0, const float* __restrict__ bhh0,
               const float* __restrict__ Wih1, const float* __restrict__ Whh1,
               const float* __restrict__ bih1, const float* __restrict__ bhh1) {
  __shared__ float xs[DECIN];
  __shared__ float h0s[HDEC], h1s[HDEC];
  const int b = blockIdx.x, j = threadIdx.x;
  for (int i = j; i < DECIN; i += 256) xs[i] = x[(size_t)b * DECIN + i];
  h0s[j] = h0_in[b * HDEC + j];
  h1s[j] = h1_in[b * HDEC + j];
  __syncthreads();
  { // layer 0
    float gr = bih0[j], gz = bih0[256 + j], gn = bih0[512 + j];
    for (int i = 0; i < DECIN; ++i) {
      float xv = xs[i];
      gr += xv * Wih0[(size_t)j * DECIN + i];
      gz += xv * Wih0[(size_t)(256 + j) * DECIN + i];
      gn += xv * Wih0[(size_t)(512 + j) * DECIN + i];
    }
    float hr = bhh0[j], hz = bhh0[256 + j], hn = bhh0[512 + j];
    for (int i = 0; i < 256; ++i) {
      float hv = h0s[i];
      hr += hv * Whh0[j * 256 + i];
      hz += hv * Whh0[(256 + j) * 256 + i];
      hn += hv * Whh0[(512 + j) * 256 + i];
    }
    float r = sigf(gr + hr), z = sigf(gz + hz);
    float n = tanhf(gn + r * hn);
    h0_out[b * HDEC + j] = (1.f - z) * n + z * h0s[j];
  }
  { // layer 1, input = OLD h0 (reference quirk)
    float gr = bih1[j], gz = bih1[256 + j], gn = bih1[512 + j];
    for (int i = 0; i < 256; ++i) {
      float xv = h0s[i];
      gr += xv * Wih1[j * 256 + i];
      gz += xv * Wih1[(256 + j) * 256 + i];
      gn += xv * Wih1[(512 + j) * 256 + i];
    }
    float hr = bhh1[j], hz = bhh1[256 + j], hn = bhh1[512 + j];
    for (int i = 0; i < 256; ++i) {
      float hv = h1s[i];
      hr += hv * Whh1[j * 256 + i];
      hz += hv * Whh1[(256 + j) * 256 + i];
      hn += hv * Whh1[(512 + j) * 256 + i];
    }
    float r = sigf(gr + hr), z = sigf(gz + hz);
    float n = tanhf(gn + r * hn);
    h1_out[b * HDEC + j] = (1.f - z) * n + z * h1s[j];
  }
}

// ---------------- host orchestration ----------------
extern "C" void kernel_launch(void* const* d_in, const int* in_sizes, int n_in,
                              void* d_out, int out_size, void* d_ws, size_t ws_size,
                              hipStream_t stream) {
  (void)in_sizes; (void)n_in; (void)out_size; (void)ws_size;
  const float* Er    = (const float*)d_in[0];
  const float* Ei    = (const float*)d_in[1];
  const float* wemb  = (const float*)d_in[2];
  const float* wlut  = (const float*)d_in[3];
  const float* gWh   = (const float*)d_in[4];
  const float* gbh   = (const float*)d_in[5];
  const float* gWt   = (const float*)d_in[6];
  const float* gbt   = (const float*)d_in[7];
  const float* gWr   = (const float*)d_in[8];
  const float* gbr   = (const float*)d_in[9];
  const float* Wgtop = (const float*)d_in[10];
  const float* Wetop = (const float*)d_in[11];
  const float* Wgh   = (const float*)d_in[12];
  const float* Wgm   = (const float*)d_in[13];
  const float* Weh   = (const float*)d_in[14];
  const float* Wem   = (const float*)d_in[15];
  const float* tW    = (const float*)d_in[16];
  const float* tb    = (const float*)d_in[17];
  const float* eWih0 = (const float*)d_in[18];
  const float* eWhh0 = (const float*)d_in[19];
  const float* ebih0 = (const float*)d_in[20];
  const float* ebhh0 = (const float*)d_in[21];
  const float* eWih1 = (const float*)d_in[22];
  const float* eWhh1 = (const float*)d_in[23];
  const float* ebih1 = (const float*)d_in[24];
  const float* ebhh1 = (const float*)d_in[25];
  const float* dWih0 = (const float*)d_in[26];
  const float* dWhh0 = (const float*)d_in[27];
  const float* dbih0 = (const float*)d_in[28];
  const float* dbhh0 = (const float*)d_in[29];
  const float* dWih1 = (const float*)d_in[30];
  const float* dWhh1 = (const float*)d_in[31];
  const float* dbih1 = (const float*)d_in[32];
  const float* dbhh1 = (const float*)d_in[33];
  const float* Wword = (const float*)d_in[34];
  float* out = (float*)d_out;

  // ---- workspace carve-out (256B aligned) ----
  char* ws = (char*)d_ws;
  size_t off = 0;
  auto carve = [&](size_t bytes) -> char* {
    char* p = ws + off;
    off = (off + bytes + 255) & ~(size_t)255;
    return p;
  };
  _Float16* Wf16   = (_Float16*)carve((size_t)VSZ * HDEC * 2);
  _Float16* hf16   = (_Float16*)carve((size_t)BB * HDEC * 2);
  float* g1        = (float*)carve((size_t)BB * SS * 200 * 4);
  float* enc_all   = (float*)carve((size_t)BB * SS * HENC * 4);
  float* g_top_    = (float*)carve((size_t)BB * SS * VG * 4);
  float* e_top_    = (float*)carve((size_t)BB * SS * VE * 4);
  float* ench0[2]  = {(float*)carve(BB * HENC * 4), (float*)carve(BB * HENC * 4)};
  float* ench1[2]  = {(float*)carve(BB * HENC * 4), (float*)carve(BB * HENC * 4)};
  float* dech0[2]  = {(float*)carve(BB * HDEC * 4), (float*)carve(BB * HDEC * 4)};
  float* dech1[2]  = {(float*)carve(BB * HDEC * 4), (float*)carve(BB * HDEC * 4)};
  float* pw        = (float*)carve(BB * WEMB * 4);
  float* pk[2]     = {(float*)carve(BB * TRIPD * 4), (float*)carve(BB * TRIPD * 4)};
  float* alpha_top = (float*)carve(BB * SS * 4);
  float* inter_re  = (float*)carve(BB * TRIPD * 4);
  float* inter_im  = (float*)carve(BB * TRIPD * 4);
  float* a_buf     = (float*)carve((size_t)BB * SS * KTR * 4);
  float* c_g       = (float*)carve(BB * 200 * 4);
  float* c_hier    = (float*)carve(BB * TRIPD * 4);
  float* c_e       = (float*)carve(BB * HENC * 4);
  float* xbuf      = (float*)carve((size_t)BB * DECIN * 4);
  float* logits    = (float*)carve((size_t)BB * VSZ * 4);
  int*   idx       = (int*)carve(BB * 4);

  const int NTILE_BLOCKS = (VSZ / 16 + 7) / 8;   // 235

  // --- setup: W_word -> f16, graph attention, zero init ---
  {
    int n = VSZ * HDEC;
    k_f32_to_f16<<<4096, 256, 0, stream>>>(Wword, Wf16, n);
  }
  k_graph_att<<<BB * SS, 128, 0, stream>>>(Er, Ei, gWh, gbh, gWt, gbt, gWr, gbr, g1);
  k_zero<<<8, 256, 0, stream>>>(ench0[0], BB * HENC);
  k_zero<<<8, 256, 0, stream>>>(ench1[0], BB * HENC);
  k_zero<<<8, 256, 0, stream>>>(pk[0], BB * TRIPD);

  // --- encoder: 50 sequential steps, ping-pong h ---
  for (int t = 0; t < SS; ++t) {
    int in = t & 1, o = (t + 1) & 1;
    k_enc_step<<<BB, 256, 0, stream>>>(wemb, g1, ench0[in], ench1[in], ench0[o], ench1[o],
                                       eWih0, eWhh0, ebih0, ebhh0,
                                       eWih1, eWhh1, ebih1, ebhh1, enc_all, t);
  }
  // final encoder states land in index 0 (50 steps: 0->1->0...)
  const float* ch0 = ench0[0];
  const float* ch1 = ench1[0];

  k_gtop<<<BB * SS, VG, 0, stream>>>(g1, Wgtop, g_top_);
  k_etop<<<BB * SS, VE, 0, stream>>>(enc_all, Wetop, e_top_);

  // --- bootstrap pw0 = word_lookup[argmax(h1f @ W_word^T)] ---
  k_f32_to_f16<<<32, 256, 0, stream>>>(ch1, hf16, BB * HDEC);
  k_logits_wmma<<<NTILE_BLOCKS, 256, 0, stream>>>(hf16, Wf16, logits);
  k_logsoftmax_argmax<<<BB, 256, 0, stream>>>(logits, (float*)nullptr, idx, 0);
  k_gather_pw<<<BB, 64, 0, stream>>>(wlut, idx, pw);

  // --- decoder: 50 sequential steps ---
  for (int t = 0; t < SS; ++t) {
    int cur = t & 1, nxt = (t + 1) & 1;
    k_dec_att_g<<<BB, 256, 0, stream>>>(ch1, g_top_, g1, Wgh, Wgm, c_g, alpha_top);
    k_inter<<<BB, 256, 0, stream>>>(ch1, tW, tb, inter_re, inter_im);
    k_beta_t<<<BB * SS, 128, 0, stream>>>(Er, Ei, inter_re, inter_im, alpha_top, a_buf);
    k_chier<<<BB, 256, 0, stream>>>(Er, Ei, a_buf, c_hier, pk[nxt]);     // pk_new
    k_dec_att_e<<<BB, 256, 0, stream>>>(ch1, e_top_, enc_all, Weh, Wem, c_e);
    k_build_x<<<BB, 256, 0, stream>>>(c_g, c_hier, pk[cur], c_e, pw, xbuf);  // old pk, old pw
    k_dec_gru<<<BB, 256, 0, stream>>>(xbuf, ch0, ch1, dech0[cur], dech1[cur],
                                      dWih0, dWhh0, dbih0, dbhh0,
                                      dWih1, dWhh1, dbih1, dbhh1);
    ch0 = dech0[cur];
    ch1 = dech1[cur];
    k_f32_to_f16<<<32, 256, 0, stream>>>(ch1, hf16, BB * HDEC);
    k_logits_wmma<<<NTILE_BLOCKS, 256, 0, stream>>>(hf16, Wf16, logits);
    k_logsoftmax_argmax<<<BB, 256, 0, stream>>>(logits, out, idx, t);
    k_gather_pw<<<BB, 64, 0, stream>>>(wlut, idx, pw);
  }
}
